// Linear_24180665876920
// MI455X (gfx1250) — 8-bit matrix operand order repaired and checked by reading, compiled, not run on hardware
//
#include <hip/hip_runtime.h>
#include <hip/hip_bf16.h>

#define M_DIM 8192
#define K_DIM 7168
#define N_DIM 4096
#define BS 128
#define KBLK (K_DIM / BS)
#define FP8_MAX 448.0f
#define AMAX_EPS 1e-4f

typedef __attribute__((ext_vector_type(16))) int   v16i;
typedef __attribute__((ext_vector_type(8)))  float v8f;
typedef int v4i __attribute__((vector_size(16)));
typedef __attribute__((address_space(1))) v4i* gptr_v4i;
typedef __attribute__((address_space(3))) v4i* lptr_v4i;

#define HAS_ASYNC_LDS __has_builtin(__builtin_amdgcn_global_load_async_to_lds_b128)

__device__ __forceinline__ unsigned int f32_to_e4m3(float f) {
    unsigned int x = __float_as_uint(f);
    unsigned int sign = (x >> 24) & 0x80u;
    x &= 0x7fffffffu;
    if (x >= 0x43e00000u) return sign | 0x7Eu;
    if (x < 0x3c800000u) {
        unsigned int m = (unsigned int)__float2int_rn(__uint_as_float(x) * 512.0f);
        return sign | m;
    }
    unsigned int y  = x + 0x7ffffu + ((x >> 20) & 1u);
    unsigned int em = y >> 20;
    int e8 = (int)(em >> 3) - 127 + 7;
    unsigned int r = ((unsigned int)e8 << 3) | (em & 7u);
    if (r > 0x7Eu) r = 0x7Eu;
    return sign | r;
}

__device__ __forceinline__ unsigned int pack4_e4m3(float a, float b, float c, float d) {
#if __has_builtin(__builtin_amdgcn_cvt_pk_fp8_f32)
    int v = 0;
    v = __builtin_amdgcn_cvt_pk_fp8_f32(a, b, v, false);
    v = __builtin_amdgcn_cvt_pk_fp8_f32(c, d, v, true);
    return (unsigned int)v;
#else
    return f32_to_e4m3(a) | (f32_to_e4m3(b) << 8) |
           (f32_to_e4m3(c) << 16) | (f32_to_e4m3(d) << 24);
#endif
}

__device__ __forceinline__ void wait_async_lds() {
#if __has_builtin(__builtin_amdgcn_s_wait_asynccnt)
    __builtin_amdgcn_s_wait_asynccnt(0);
#else
    asm volatile("s_wait_asynccnt 0x0" ::: "memory");
#endif
}

__global__ __launch_bounds__(256)
void quant_x_kernel(const __hip_bfloat16* __restrict__ x,
                    unsigned char* __restrict__ xq,
                    float* __restrict__ sx) {
    const int lane = threadIdx.x & 31;
    const int wave = blockIdx.x * (blockDim.x >> 5) + (threadIdx.x >> 5);
    const int m  = wave / KBLK;
    const int kb = wave - m * KBLK;
    const size_t base = (size_t)m * K_DIM + (size_t)kb * BS;

    const unsigned int* src = (const unsigned int*)((const unsigned short*)x + base) + lane * 2;
    const unsigned int r0 = src[0], r1 = src[1];
    const float f0 = __uint_as_float(r0 << 16);
    const float f1 = __uint_as_float(r0 & 0xffff0000u);
    const float f2 = __uint_as_float(r1 << 16);
    const float f3 = __uint_as_float(r1 & 0xffff0000u);

    float a = fmaxf(fmaxf(fabsf(f0), fabsf(f1)), fmaxf(fabsf(f2), fabsf(f3)));
    #pragma unroll
    for (int off = 16; off > 0; off >>= 1) a = fmaxf(a, __shfl_xor(a, off, 32));
    a = fmaxf(a, AMAX_EPS);
    const float s   = a * (1.0f / FP8_MAX);
    const float inv = FP8_MAX / a;

    const float q0 = fminf(fmaxf(f0 * inv, -FP8_MAX), FP8_MAX);
    const float q1 = fminf(fmaxf(f1 * inv, -FP8_MAX), FP8_MAX);
    const float q2 = fminf(fmaxf(f2 * inv, -FP8_MAX), FP8_MAX);
    const float q3 = fminf(fmaxf(f3 * inv, -FP8_MAX), FP8_MAX);

    ((unsigned int*)xq)[base / 4 + lane] = pack4_e4m3(q0, q1, q2, q3);
    if (lane == 0) sx[m * KBLK + kb] = s;
}

__global__ __launch_bounds__(256)
void quant_w_kernel(const float* __restrict__ w, unsigned char* __restrict__ wq8) {
    const size_t t = (size_t)blockIdx.x * blockDim.x + threadIdx.x;
    const float4 v = ((const float4*)w)[t];
    ((unsigned int*)wq8)[t] = pack4_e4m3(v.x, v.y, v.z, v.w);
}

__global__ __launch_bounds__(256)
void fp8_gemm_kernel(const unsigned char* __restrict__ xq,
                     const float* __restrict__ sx,
                     const unsigned char* __restrict__ wq8,
                     const float* __restrict__ ws,
                     __hip_bfloat16* __restrict__ out) {
    __shared__ int4 lBv[2 * BS * BS / 16];
    unsigned char* lB = (unsigned char*)lBv;

    const int lane = threadIdx.x & 31;
    const int wv   = threadIdx.x >> 5;
    const int n0   = blockIdx.x * BS;
    const int m0   = blockIdx.y * BS;
    const int mw   = m0 + wv * 16;
    const int nb   = blockIdx.x;

    const int nl    = lane & 15;
    const int hi    = lane >> 4;
    const unsigned char* aRow = xq + (size_t)(mw + nl) * K_DIM + hi * 8;
    const float* sRow = sx + (size_t)(mw + hi * 8) * KBLK;
    const int khalf = hi * 16;

    const int sn = threadIdx.x >> 1;
    const int sh = (threadIdx.x & 1) * 64;
    const unsigned char* gW   = wq8 + (size_t)(n0 + sn) * K_DIM + sh;
    unsigned char*       lDst = lB + sn * BS + sh;

    float acc[8][8];
    #pragma unroll
    for (int t = 0; t < 8; ++t)
        #pragma unroll
        for (int j = 0; j < 8; ++j) acc[t][j] = 0.0f;

    auto stage = [&](int kb, int buf) {
        const unsigned char* gsrc = gW + (size_t)kb * BS;
        unsigned char*       ldst = lDst + buf * (BS * BS);
#if HAS_ASYNC_LDS
        #pragma unroll
        for (int i = 0; i < 4; ++i)
            __builtin_amdgcn_global_load_async_to_lds_b128(
                (gptr_v4i)(gsrc + i * 16),
                (lptr_v4i)(ldst + i * 16),
                0, 0);
#else
        #pragma unroll
        for (int i = 0; i < 4; ++i)
            ((int4*)ldst)[i] = ((const int4*)gsrc)[i];
#endif
    };

    stage(0, 0);

    for (int kb = 0; kb < KBLK; ++kb) {
        const int p = kb & 1;
#if HAS_ASYNC_LDS
        wait_async_lds();
#endif
        __syncthreads();
        if (kb + 1 < KBLK) stage(kb + 1, p ^ 1);

        union { v16i v; unsigned long long d[8]; } A;
        const unsigned char* ak = aRow + (size_t)kb * BS;
        #pragma unroll
        for (int qd = 0; qd < 8; ++qd)
            A.d[qd] = *(const unsigned long long*)(ak + (qd >> 1) * 32 + (qd & 1) * 8 + hi * 8);

        const float wsv = ws[nb * KBLK + kb];
        float sj[8];
        #pragma unroll
        for (int j = 0; j < 8; ++j) sj[j] = sRow[j * KBLK + kb] * wsv;

        const unsigned char* lCur = lB + p * (BS * BS);
        #pragma unroll
        for (int t = 0; t < 8; ++t) {
            union { v16i v; int4 q[4]; } B;
            const unsigned char* bk = lCur + (t * 16 + nl) * BS + khalf;
            #pragma unroll
            for (int g = 0; g < 4; ++g) B.q[g] = *(const int4*)(bk + g * 32);
            v8f c0 = {};
            v8f d = __builtin_amdgcn_wmma_f32_16x16x128_fp8_fp8(A.v, B.v, (short)0, c0,
                                                                false, false);
            #pragma unroll
            for (int j = 0; j < 8; ++j) acc[t][j] = fmaf(d[j], sj[j], acc[t][j]);
        }
        __syncthreads();
    }

    #pragma unroll
    for (int t = 0; t < 8; ++t) {
        const int n = n0 + t * 16 + nl;
        #pragma unroll
        for (int j = 0; j < 8; ++j) {
            const int m = mw + hi * 8 + j;
            out[(size_t)m * N_DIM + n] = __float2bfloat16(acc[t][j]);
        }
    }
}

extern "C" void kernel_launch(void* const* d_in, const int* in_sizes, int n_in,
                              void* d_out, int out_size, void* d_ws, size_t ws_size,
                              hipStream_t stream) {
    (void)in_sizes; (void)n_in; (void)out_size; (void)ws_size;
    const __hip_bfloat16* x   = (const __hip_bfloat16*)d_in[0];
    const float*          w_q = (const float*)d_in[1];
    const float*          w_s = (const float*)d_in[2];
    __hip_bfloat16*       out = (__hip_bfloat16*)d_out;

    unsigned char* xq  = (unsigned char*)d_ws;
    unsigned char* wq8 = xq + (size_t)M_DIM * K_DIM;
    float*         sx  = (float*)(wq8 + (size_t)N_DIM * K_DIM);

    quant_x_kernel<<<(M_DIM * KBLK) / 8, 256, 0, stream>>>(x, xq, sx);
    quant_w_kernel<<<(int)((size_t)N_DIM * K_DIM / 4 / 256), 256, 0, stream>>>(w_q, wq8);

    dim3 grid(N_DIM / BS, M_DIM / BS);
    fp8_gemm_kernel<<<grid, 256, 0, stream>>>(xq, sx, wq8, w_s, out);
}
